// RQwenAttention_45792941310596
// MI455X (gfx1250) — matrix operand order repaired and checked by reading, not compiled, not run on hardware
//
#include <hip/hip_runtime.h>
#include <hip/hip_bf16.h>

typedef __attribute__((ext_vector_type(16))) __bf16 v16bf;
typedef __attribute__((ext_vector_type(8)))  float  v8f;

union Frag16 { uint4 u4[2]; v16bf v; };

__device__ __forceinline__ v8f wmma_bf16(v16bf a, v16bf b, v8f c) {
  return __builtin_amdgcn_wmma_f32_16x16x32_bf16(false, a, false, b, (short)0, c, false, false);
}

// Async copy 16B global -> LDS (CDNA5 TDM-lite path, tracked by ASYNCcnt).
// lds = 32-bit LDS byte offset (low dword of generic shared pointer), g = global addr.
__device__ __forceinline__ void async_cp16(unsigned lds, const void* g) {
  asm volatile("global_load_async_to_lds_b128 %0, %1, off" :: "v"(lds), "v"(g) : "memory");
}

// A-fragment (16x32 MxK, 16-bit) from row-major [M][K] LDS tile.
__device__ __forceinline__ v16bf load_a_frag(const __bf16* base, int stride, int lane) {
  int m = lane & 15;
  int koff = (lane >> 4) << 3;
  Frag16 f;
  f.u4[0] = *(const uint4*)(base + m * stride + koff);
  f.u4[1] = *(const uint4*)(base + m * stride + 16 + koff);
  return f.v;
}

// B-fragment (32x16 KxN, 16-bit) from LDS tile stored [N][K] row-major.
__device__ __forceinline__ v16bf load_b_frag(const __bf16* base, int stride, int lane) {
  int n = lane & 15;
  int kb = (lane >> 4) << 3;
  const __bf16* p = base + n * stride + kb;
  Frag16 f;
  f.u4[0] = *(const uint4*)(p);
  f.u4[1] = *(const uint4*)(p + 16);
  return f.v;
}

// ---------------------------------------------------------------- converts
__global__ void cvt_f32_bf16(const float* __restrict__ in, __bf16* __restrict__ out, int n) {
  int idx = blockIdx.x * blockDim.x + threadIdx.x;
  if (idx < n) out[idx] = (__bf16)in[idx];
}

// ---------------------------------------------------------------- RoPE (in-place, bf16 [B,S,H,D])
__global__ void rope_bf16(__bf16* __restrict__ q, __bf16* __restrict__ k) {
  int idx = blockIdx.x * blockDim.x + threadIdx.x;  // B*S*H*64 threads
  int i  = idx & 63;
  int h  = (idx >> 6) & 15;
  int bs = idx >> 10;            // b*2048 + s
  int s  = bs & 2047;
  size_t base = (((size_t)bs * 16) + h) * 128;
  const float LT = 0.14391156831212787f;   // ln(10000)/64
  float ang = (float)s * __expf(-(float)i * LT);
  float cs = __cosf(ang), sn = __sinf(ang);
  float q1 = (float)q[base + i], q2 = (float)q[base + i + 64];
  q[base + i]      = (__bf16)(q1 * cs - q2 * sn);
  q[base + i + 64] = (__bf16)(q2 * cs + q1 * sn);
  float k1 = (float)k[base + i], k2 = (float)k[base + i + 64];
  k[base + i]      = (__bf16)(k1 * cs - k2 * sn);
  k[base + i + 64] = (__bf16)(k2 * cs + k1 * sn);
}

// ---------------------------------------------------------------- generic RLoRA GEMM
// C[M,N] = A[M,K] * W[N,K]^T (+ bias) (+ A2[M,K2] * W2[N,K2]^T)
// EPI: 0 = store f32, 1 = store bf16, 2 = store bf16 * expert_mask (N==128 LoRA 'a')
#define GTM 128
#define GTN 128
#define GTK 64
#define GSTR 72   // 64 + 8 pad (bf16 elems)

template<int EPI, bool LORA, bool BIAS>
__global__ __launch_bounds__(256) void gemm_rlora(
    const __bf16* __restrict__ A,  const __bf16* __restrict__ W,
    const float*  __restrict__ bias,
    const __bf16* __restrict__ A2, const __bf16* __restrict__ W2,
    const float*  __restrict__ mask,
    float* __restrict__ outf, __bf16* __restrict__ outb,
    int M, int N, int K, int K2)
{
  __shared__ __align__(16) __bf16 As[2][GTM * GSTR];
  __shared__ __align__(16) __bf16 Bs[2][GTN * GSTR];

  const int tid   = threadIdx.x;
  const int lane  = tid & 31;
  const int wid   = tid >> 5;
  const int waveM = wid >> 1;      // 0..3
  const int waveN = wid & 1;       // 0..1
  const int m0 = blockIdx.y * GTM;
  const int n0 = blockIdx.x * GTN;

  v8f zero8 = {0.f,0.f,0.f,0.f,0.f,0.f,0.f,0.f};
  v8f c[2][4];
#pragma unroll
  for (int i = 0; i < 2; ++i)
#pragma unroll
    for (int j = 0; j < 4; ++j) c[i][j] = zero8;

  const int kt = K >> 6;
  const int ntiles = kt + (LORA ? (K2 >> 6) : 0);
  const int row  = tid >> 1;
  const int part = tid & 1;

  // Issue one stage of async global->LDS copies (8 x b128 per thread).
  auto stage_async = [&](int buf, int t) {
    const __bf16* Ag; const __bf16* Wg; int k0, ld;
    if (!LORA || t < kt) { Ag = A;  Wg = W;  k0 = t * GTK;        ld = K;  }
    else                 { Ag = A2; Wg = W2; k0 = (t - kt) * GTK; ld = K2; }
    const __bf16* ga = Ag + (size_t)(m0 + row) * ld + k0 + part * 32;
    const __bf16* gb = Wg + (size_t)(n0 + row) * ld + k0 + part * 32;
    unsigned la = (unsigned)(uintptr_t)(&As[buf][row * GSTR + part * 32]);
    unsigned lb = (unsigned)(uintptr_t)(&Bs[buf][row * GSTR + part * 32]);
    async_cp16(la +  0, ga +  0);
    async_cp16(la + 16, ga +  8);
    async_cp16(la + 32, ga + 16);
    async_cp16(la + 48, ga + 24);
    async_cp16(lb +  0, gb +  0);
    async_cp16(lb + 16, gb +  8);
    async_cp16(lb + 32, gb + 16);
    async_cp16(lb + 48, gb + 24);
  };

  auto mma_step = [&](int buf) {
#pragma unroll
    for (int kk = 0; kk < 2; ++kk) {
      v16bf af[2], bf[4];
#pragma unroll
      for (int i = 0; i < 2; ++i)
        af[i] = load_a_frag(&As[buf][(waveM * 32 + i * 16) * GSTR + kk * 32], GSTR, lane);
#pragma unroll
      for (int j = 0; j < 4; ++j)
        bf[j] = load_b_frag(&Bs[buf][(waveN * 64 + j * 16) * GSTR + kk * 32], GSTR, lane);
#pragma unroll
      for (int i = 0; i < 2; ++i)
#pragma unroll
        for (int j = 0; j < 4; ++j)
          c[i][j] = wmma_bf16(af[i], bf[j], c[i][j]);
    }
  };

  // Double-buffered pipeline: tile t+1 streams into buf^1 while tile t computes.
  stage_async(0, 0);
  for (int t = 0; t < ntiles; ++t) {
    int cur = t & 1;
    if (t + 1 < ntiles) {
      stage_async(cur ^ 1, t + 1);
      asm volatile("s_wait_asynccnt 0x8" ::: "memory");  // tile t's 8 ops done
    } else {
      asm volatile("s_wait_asynccnt 0x0" ::: "memory");
    }
    __syncthreads();      // all waves' tile-t data visible
    mma_step(cur);
    __syncthreads();      // tile-t buffer free for overwrite at t+1's issue
  }

#pragma unroll
  for (int i = 0; i < 2; ++i)
#pragma unroll
    for (int j = 0; j < 4; ++j)
#pragma unroll
      for (int e = 0; e < 8; ++e) {
        int rowo = m0 + waveM * 32 + i * 16 + e + ((lane >> 4) << 3);
        int col  = n0 + waveN * 64 + j * 16 + (lane & 15);
        float v = c[i][j][e];
        if constexpr (BIAS) v += bias[col];
        if constexpr (EPI == 0)      outf[(size_t)rowo * N + col] = v;
        else if constexpr (EPI == 1) outb[(size_t)rowo * N + col] = (__bf16)v;
        else                         outb[(size_t)rowo * N + col] =
                                       (__bf16)(v * mask[rowo * 8 + (col >> 4)]);
      }
}

// ---------------------------------------------------------------- flash attention
#define KSTR 136   // 128 + 8
#define VSTR 72    // 64 + 8
#define PSTR 72

__global__ __launch_bounds__(128) void attn_fa(
    const __bf16* __restrict__ qb, const __bf16* __restrict__ kb,
    const __bf16* __restrict__ vb, __bf16* __restrict__ ctxb)
{
  __shared__ __align__(16) __bf16 Klds[64 * KSTR];       // [k'][d] == [N][K] for QK^T
  __shared__ __align__(16) __bf16 Vlds[128 * VSTR];      // [d][k'] == [N][K] for P.V
  __shared__ __align__(16) __bf16 Plds[4 * 16 * PSTR];   // per-wave P [16][64]

  const int tid  = threadIdx.x;
  const int lane = tid & 31;
  const int wid  = tid >> 5;
  const int bh = blockIdx.x;
  const int b = bh >> 4, h = bh & 15;
  const int q0 = blockIdx.y * 64;
  const int qrow0 = q0 + wid * 16;

  // Q fragments for this wave's 16 rows: 4 chunks of K(=d)=32
  v16bf qf[4];
  {
    int m = lane & 15;
    int koff = (lane >> 4) << 3;
    const __bf16* qrow = qb + (((size_t)(b * 2048 + qrow0 + m)) * 16 + h) * 128;
#pragma unroll
    for (int kk = 0; kk < 4; ++kk) {
      Frag16 f;
      f.u4[0] = *(const uint4*)(qrow + kk * 32 + koff);
      f.u4[1] = *(const uint4*)(qrow + kk * 32 + 16 + koff);
      qf[kk] = f.v;
    }
  }

  v8f zero8 = {0.f,0.f,0.f,0.f,0.f,0.f,0.f,0.f};
  float mrun[8], lrun[8];
  v8f ctx[8];
#pragma unroll
  for (int e = 0; e < 8; ++e) { mrun[e] = -3.0e30f; lrun[e] = 0.f; }
#pragma unroll
  for (int i = 0; i < 8; ++i) ctx[i] = zero8;

  const int jmax = q0 >> 6;
  for (int jb = 0; jb <= jmax; ++jb) {
    const int kb0 = jb * 64;
    __syncthreads();
    { // stage K tile [64][128] row-major via async-to-LDS
      int row = tid >> 1, part = tid & 1;
      const __bf16* src = kb + (((size_t)(b * 2048 + kb0 + row)) * 16 + h) * 128 + part * 64;
      unsigned la = (unsigned)(uintptr_t)(&Klds[row * KSTR + part * 64]);
#pragma unroll
      for (int u = 0; u < 8; ++u)
        async_cp16(la + u * 16, src + u * 8);
    }
    { // stage V transposed: Vlds[d][kp]
      int kp = tid >> 1, part = tid & 1;
      const __bf16* src = vb + (((size_t)(b * 2048 + kb0 + kp)) * 16 + h) * 128 + part * 64;
#pragma unroll
      for (int d = 0; d < 64; ++d)
        Vlds[(part * 64 + d) * VSTR + kp] = src[d];
    }
    // pull next block toward L2 while this one computes
    if (jb < jmax) {
      __builtin_prefetch(kb + (((size_t)(b * 2048 + kb0 + 64 + (tid >> 1))) * 16 + h) * 128, 0, 1);
      __builtin_prefetch(vb + (((size_t)(b * 2048 + kb0 + 64 + (tid >> 1))) * 16 + h) * 128, 0, 1);
    }
    asm volatile("s_wait_asynccnt 0x0" ::: "memory");
    __syncthreads();

    // scores S = Q K^T (16x64 per wave)
    v8f s[4];
#pragma unroll
    for (int j = 0; j < 4; ++j) s[j] = zero8;
#pragma unroll
    for (int j = 0; j < 4; ++j)
#pragma unroll
      for (int kk = 0; kk < 4; ++kk)
        s[j] = wmma_bf16(qf[kk], load_b_frag(Klds + j * 16 * KSTR + kk * 32, KSTR, lane), s[j]);

    const float SC = 0.08838834764831845f;   // 1/sqrt(128)
    int rbase = qrow0 + ((lane >> 4) << 3);
    int cbase = kb0 + (lane & 15);
#pragma unroll
    for (int e = 0; e < 8; ++e) {
      int row = rbase + e;
      float tm = -3.0e30f;
#pragma unroll
      for (int j = 0; j < 4; ++j) {
        float val = s[j][e] * SC;
        val = (cbase + j * 16 <= row) ? val : -3.0e30f;   // causal
        s[j][e] = val;
        tm = fmaxf(tm, val);
      }
#pragma unroll
      for (int off = 8; off >= 1; off >>= 1)
        tm = fmaxf(tm, __shfl_xor(tm, off, 32));
      float mnew = fmaxf(mrun[e], tm);
      float corr = __expf(mrun[e] - mnew);
      mrun[e] = mnew;
      float sum = 0.f;
#pragma unroll
      for (int j = 0; j < 4; ++j) {
        float p = __expf(s[j][e] - mnew);
        s[j][e] = p;
        sum += p;
      }
#pragma unroll
      for (int off = 8; off >= 1; off >>= 1)
        sum += __shfl_xor(sum, off, 32);
      lrun[e] = lrun[e] * corr + sum;
#pragma unroll
      for (int i = 0; i < 8; ++i) ctx[i][e] *= corr;
    }

    // spill P (bf16) to this wave's LDS region, reload as A-fragments
    __bf16* pl = Plds + wid * 16 * PSTR;
#pragma unroll
    for (int e = 0; e < 8; ++e) {
      int rl = e + ((lane >> 4) << 3);
#pragma unroll
      for (int j = 0; j < 4; ++j)
        pl[rl * PSTR + j * 16 + (lane & 15)] = (__bf16)s[j][e];
    }
    asm volatile("s_wait_dscnt 0x0" ::: "memory");

#pragma unroll
    for (int kk = 0; kk < 2; ++kk) {
      v16bf pf = load_a_frag(pl + kk * 32, PSTR, lane);
#pragma unroll
      for (int i = 0; i < 8; ++i)
        ctx[i] = wmma_bf16(pf, load_b_frag(Vlds + i * 16 * VSTR + kk * 32, VSTR, lane), ctx[i]);
    }
  }

  // normalize and store ctx as bf16 [B*S][H*D]
#pragma unroll
  for (int i = 0; i < 8; ++i)
#pragma unroll
    for (int e = 0; e < 8; ++e) {
      int row = qrow0 + e + ((lane >> 4) << 3);
      int col = h * 128 + i * 16 + (lane & 15);
      ctxb[((size_t)(b * 2048 + row)) * 2048 + col] = (__bf16)(ctx[i][e] / lrun[e]);
    }
}

// ---------------------------------------------------------------- host
extern "C" void kernel_launch(void* const* d_in, const int* in_sizes, int n_in,
                              void* d_out, int out_size, void* d_ws, size_t ws_size,
                              hipStream_t stream) {
  const float* x    = (const float*)d_in[0];
  const float* mask = (const float*)d_in[1];
  const float* Wq = (const float*)d_in[2];
  const float* bq = (const float*)d_in[3];
  const float* Wk = (const float*)d_in[4];
  const float* bk = (const float*)d_in[5];
  const float* Wv = (const float*)d_in[6];
  const float* bv = (const float*)d_in[7];
  const float* Wo = (const float*)d_in[8];
  const float* la_q = (const float*)d_in[9];
  const float* lb_q = (const float*)d_in[10];
  const float* la_k = (const float*)d_in[11];
  const float* lb_k = (const float*)d_in[12];
  const float* la_v = (const float*)d_in[13];
  const float* lb_v = (const float*)d_in[14];
  const float* la_o = (const float*)d_in[15];
  const float* lb_o = (const float*)d_in[16];
  float* out = (float*)d_out;

  const size_t MS = 4096;          // B*S
  const size_t E  = 2048;

  char* ws = (char*)d_ws;
  size_t off = 0;
  auto alloc = [&](size_t bytes) {
    void* p = ws + off;
    off += (bytes + 255) & ~(size_t)255;
    return p;
  };
  __bf16* xb   = (__bf16*)alloc(MS * E * 2);
  __bf16* Wqb  = (__bf16*)alloc(E * E * 2);
  __bf16* Wkb  = (__bf16*)alloc(E * E * 2);
  __bf16* Wvb  = (__bf16*)alloc(E * E * 2);
  __bf16* Wob  = (__bf16*)alloc(E * E * 2);
  __bf16* laqb = (__bf16*)alloc(128 * E * 2);
  __bf16* lakb = (__bf16*)alloc(128 * E * 2);
  __bf16* lavb = (__bf16*)alloc(128 * E * 2);
  __bf16* laob = (__bf16*)alloc(128 * E * 2);
  __bf16* lbqb = (__bf16*)alloc(E * 128 * 2);
  __bf16* lbkb = (__bf16*)alloc(E * 128 * 2);
  __bf16* lbvb = (__bf16*)alloc(E * 128 * 2);
  __bf16* lbob = (__bf16*)alloc(E * 128 * 2);
  __bf16* abuf = (__bf16*)alloc(MS * 128 * 2);
  __bf16* qbuf = (__bf16*)alloc(MS * E * 2);
  __bf16* kbuf = (__bf16*)alloc(MS * E * 2);
  __bf16* vbuf = (__bf16*)alloc(MS * E * 2);
  __bf16* cxb  = (__bf16*)alloc(MS * E * 2);

  auto cvt = [&](const float* src, __bf16* dst, int n) {
    cvt_f32_bf16<<<(n + 255) / 256, 256, 0, stream>>>(src, dst, n);
  };
  cvt(x,  xb,  (int)(MS * E));
  cvt(Wq, Wqb, (int)(E * E));  cvt(Wk, Wkb, (int)(E * E));
  cvt(Wv, Wvb, (int)(E * E));  cvt(Wo, Wob, (int)(E * E));
  cvt(la_q, laqb, (int)(128 * E));  cvt(la_k, lakb, (int)(128 * E));
  cvt(la_v, lavb, (int)(128 * E));  cvt(la_o, laob, (int)(128 * E));
  cvt(lb_q, lbqb, (int)(E * 128));  cvt(lb_k, lbkb, (int)(E * 128));
  cvt(lb_v, lbvb, (int)(E * 128));  cvt(lb_o, lbob, (int)(E * 128));

  dim3 g_lora(1, 32);    // N=128,  M=4096
  dim3 g_proj(16, 32);   // N=2048, M=4096

  // Q
  gemm_rlora<2, false, false><<<g_lora, 256, 0, stream>>>(
      xb, laqb, nullptr, nullptr, nullptr, mask, nullptr, abuf, 4096, 128, 2048, 0);
  gemm_rlora<1, true, true><<<g_proj, 256, 0, stream>>>(
      xb, Wqb, bq, abuf, lbqb, nullptr, nullptr, qbuf, 4096, 2048, 2048, 128);
  // K
  gemm_rlora<2, false, false><<<g_lora, 256, 0, stream>>>(
      xb, lakb, nullptr, nullptr, nullptr, mask, nullptr, abuf, 4096, 128, 2048, 0);
  gemm_rlora<1, true, true><<<g_proj, 256, 0, stream>>>(
      xb, Wkb, bk, abuf, lbkb, nullptr, nullptr, kbuf, 4096, 2048, 2048, 128);
  // V
  gemm_rlora<2, false, false><<<g_lora, 256, 0, stream>>>(
      xb, lavb, nullptr, nullptr, nullptr, mask, nullptr, abuf, 4096, 128, 2048, 0);
  gemm_rlora<1, true, true><<<g_proj, 256, 0, stream>>>(
      xb, Wvb, bv, abuf, lbvb, nullptr, nullptr, vbuf, 4096, 2048, 2048, 128);

  // RoPE on q,k (in place)
  rope_bf16<<<16384, 256, 0, stream>>>(qbuf, kbuf);

  // attention: grid (B*H, S/64)
  attn_fa<<<dim3(32, 32), 128, 0, stream>>>(qbuf, kbuf, vbuf, cxb);

  // output projection
  gemm_rlora<2, false, false><<<g_lora, 256, 0, stream>>>(
      cxb, laob, nullptr, nullptr, nullptr, mask, nullptr, abuf, 4096, 128, 2048, 0);
  gemm_rlora<0, true, false><<<g_proj, 256, 0, stream>>>(
      cxb, Wob, nullptr, abuf, lbob, nullptr, out, nullptr, 4096, 2048, 2048, 128);
}
